// AutoregressiveGraphTransformer_89790586290221
// MI455X (gfx1250) — matrix operand order repaired and checked by reading, not compiled, not run on hardware
//
#include <hip/hip_runtime.h>
#include <hip/hip_bf16.h>
#include <math.h>

// ---------------- problem constants (fixed by the reference) ----------------
#define NL    6        // layers
#define DIM   128      // hidden dim
#define NH    8        // heads
#define CH    16       // per-head dim
#define NF    9        // node features
#define NOUT  3        // output dim
#define BB_   2
#define SS_   128
#define NNODE 64
#define NTOT  (BB_ * SS_ * NNODE)   // 16384
#define NE    262144

// per-layer packed weight region: tN(8) x mat(4) x kt(4) x lane(32) x e(16) f16
#define WFRAG_PER_TN    (4 * 4 * 32 * 16)       // 8192 f16 = 16 KB
#define WFRAG_PER_LAYER (8 * WFRAG_PER_TN)      // 65536 f16

typedef __attribute__((ext_vector_type(16))) _Float16 v16h;
typedef __attribute__((ext_vector_type(8)))  float    v8f;

// ---------------------------------------------------------------------------
// 1) input projection: h = x @ Win + bin + pe[timestep]
// ---------------------------------------------------------------------------
__global__ void input_proj_kernel(const float* __restrict__ x,
                                  const float* __restrict__ pe,
                                  const float* __restrict__ Win,
                                  const float* __restrict__ bin,
                                  float* __restrict__ h) {
  int tid = blockIdx.x * blockDim.x + threadIdx.x;      // N*D threads
  if (tid >= NTOT * DIM) return;
  int n = tid >> 7;
  int d = tid & 127;
  int s = (n >> 6) & 127;                                // (n / NN) % S
  float acc = bin[d] + pe[s * DIM + d];
#pragma unroll
  for (int f = 0; f < NF; ++f)
    acc += x[n * NF + f] * Win[f * DIM + d];
  h[tid] = acc;
}

// ---------------------------------------------------------------------------
// 2a) pack h (row-major f32) into WMMA A-fragment order f16:
//     h16[ ((tM*4 + kt)*32 + lane)*16 + e ]  (ISA 7.12.2 16-bit A 16x32)
// ---------------------------------------------------------------------------
__global__ void pack_h16_kernel(const float* __restrict__ h,
                                _Float16* __restrict__ h16) {
  int tid = blockIdx.x * blockDim.x + threadIdx.x;      // N*D threads
  if (tid >= NTOT * DIM) return;
  int e    = tid & 15;
  int lane = (tid >> 4) & 31;
  int kt   = (tid >> 9) & 3;
  int tM   = tid >> 11;
  int row  = tM * 16 + (lane & 15);
  int kk   = ((e >> 3) << 4) | ((lane >> 4) << 3) | (e & 7);
  h16[tid] = (_Float16)h[row * DIM + kt * 32 + kk];
}

// ---------------------------------------------------------------------------
// 2b) pack all layers' Q/K/V/skip weights into B-fragment order f16 (once):
//     w16[ l ][ tN ][ mat ][ kt ][ lane ][ e ]
//     B 32x16 f16 layout: col = lane&15, K = (e>=8 ? 16 : 0) + (lane>=16 ? 8 : 0) + (e&7)
// ---------------------------------------------------------------------------
__global__ void pack_w16_kernel(const float* __restrict__ Wq,
                                const float* __restrict__ Wk,
                                const float* __restrict__ Wv,
                                const float* __restrict__ Ws,
                                _Float16* __restrict__ w16) {
  int tid = blockIdx.x * blockDim.x + threadIdx.x;      // NL*65536 threads
  if (tid >= NL * WFRAG_PER_LAYER) return;
  int e    = tid & 15;
  int lane = (tid >> 4) & 31;
  int kt   = (tid >> 9) & 3;
  int mat  = (tid >> 11) & 3;
  int tN   = (tid >> 13) & 7;
  int l    = tid >> 16;
  int col  = tN * 16 + (lane & 15);
  int k    = kt * 32 + ((e >> 3) << 4) + ((lane >> 4) << 3) + (e & 7);
  const float* W = (mat == 0) ? Wq : (mat == 1) ? Wk : (mat == 2) ? Wv : Ws;
  w16[tid] = (_Float16)W[(size_t)l * DIM * DIM + k * DIM + col];
}

// ---------------------------------------------------------------------------
// 2c) fused Q/K/V/skip projection with WMMA.
//     Block = 256 threads (8 waves). Block owns 8 row-tiles x 1 col-tile.
//     The 16 KB of B fragments for this col-tile are staged into LDS with
//     CDNA5 async-to-LDS loads (ASYNCcnt), then each wave issues
//     16 v_wmma_f32_16x16x32_f16 (4 K-steps x 4 weight matrices).
// ---------------------------------------------------------------------------
__global__ void qkvs_gemm_wmma(const _Float16* __restrict__ h16,
                               const _Float16* __restrict__ w16l,  // layer base
                               const float* __restrict__ bq, const float* __restrict__ bk,
                               const float* __restrict__ bv, const float* __restrict__ bs,
                               float* __restrict__ q, float* __restrict__ k,
                               float* __restrict__ v, float* __restrict__ xr) {
  __shared__ _Float16 lw[WFRAG_PER_TN];   // [mat][kt][lane][e] = 16 KB

  const int t    = threadIdx.x;
  const int lane = t & 31;
  const int wave = t >> 5;
  const int bM   = blockIdx.x >> 3;       // 0..127
  const int tN   = blockIdx.x & 7;        // 0..7
  const int tM   = bM * 8 + wave;         // 0..1023

  // ---- async stage of this col-tile's packed weights: 16 KB, 16 B per lane x4
  {
    const char* wsrc = (const char*)(w16l + (size_t)tN * WFRAG_PER_TN);
    unsigned lds_base = (unsigned)(uintptr_t)&lw[0];
#pragma unroll
    for (int i = 0; i < 4; ++i) {
      unsigned  loff = lds_base + (unsigned)(t + i * 256) * 16u;
      const void* gp = wsrc + (size_t)(t + i * 256) * 16u;
      asm volatile("global_load_async_to_lds_b128 %0, %1, off"
                   :: "v"(loff), "v"(gp) : "memory");
    }
    asm volatile("s_wait_asynccnt 0x0" ::: "memory");
  }
  __syncthreads();

  const v16h* hv = (const v16h*)h16;      // [ (tM*4+kt)*32 + lane ]
  const v16h* lv = (const v16h*)lw;       // [ (mat*4+kt)*32 + lane ]

  v8f cq = {}, ck = {}, cv = {}, cs = {};
#pragma unroll
  for (int kt = 0; kt < 4; ++kt) {
    v16h a  = hv[(tM * 4 + kt) * 32 + lane];
    v16h b0 = lv[(0 * 4 + kt) * 32 + lane];
    v16h b1 = lv[(1 * 4 + kt) * 32 + lane];
    v16h b2 = lv[(2 * 4 + kt) * 32 + lane];
    v16h b3 = lv[(3 * 4 + kt) * 32 + lane];
    cq = __builtin_amdgcn_wmma_f32_16x16x32_f16(false, a, false, b0, (short)0, cq, false, false);
    ck = __builtin_amdgcn_wmma_f32_16x16x32_f16(false, a, false, b1, (short)0, ck, false, false);
    cv = __builtin_amdgcn_wmma_f32_16x16x32_f16(false, a, false, b2, (short)0, cv, false, false);
    cs = __builtin_amdgcn_wmma_f32_16x16x32_f16(false, a, false, b3, (short)0, cs, false, false);
  }

  // C/D layout: VGPR rr -> M = rr + (lane>=16 ? 8 : 0), N = lane&15
  const int r   = lane & 15;
  const int hi  = (lane >= 16) ? 8 : 0;
  const int c   = tN * 16 + r;
  const int row0 = tM * 16;
#pragma unroll
  for (int rr = 0; rr < 8; ++rr) {
    const int m = row0 + hi + rr;
    q[m * DIM + c]  = cq[rr] + bq[c];
    k[m * DIM + c]  = ck[rr] + bk[c];
    v[m * DIM + c]  = cv[rr] + bv[c];
    xr[m * DIM + c] = cs[rr] + bs[c];
  }
}

// ---------------------------------------------------------------------------
// 3) per-layer scratch init: agg = 0, m = -inf, denom = 0
// ---------------------------------------------------------------------------
__global__ void layer_init_kernel(float* __restrict__ agg,
                                  float* __restrict__ m,
                                  float* __restrict__ den) {
  int tid = blockIdx.x * blockDim.x + threadIdx.x;      // N*D threads
  if (tid >= NTOT * DIM) return;
  agg[tid] = 0.0f;
  if (tid < NTOT * NH) { m[tid] = -INFINITY; den[tid] = 0.0f; }
}

__device__ inline void atomicMaxF(float* addr, float val) {
  unsigned int* ai = (unsigned int*)addr;
  unsigned int old = *ai;
  for (;;) {
    float cur = __uint_as_float(old);
    if (cur >= val) break;
    unsigned int assumed = old;
    old = atomicCAS(ai, assumed, __float_as_uint(val));
    if (old == assumed) break;
  }
}

// ---------------------------------------------------------------------------
// 4) edge scores (q[dst]·k[src] per head) + fused segment-max
// ---------------------------------------------------------------------------
__global__ void edge_score_kernel(const int* __restrict__ ei,
                                  const float* __restrict__ q,
                                  const float* __restrict__ k,
                                  float* __restrict__ sc,
                                  float* __restrict__ m) {
  int tid = blockIdx.x * blockDim.x + threadIdx.x;      // E*H threads
  if (tid >= NE * NH) return;
  int e  = tid >> 3;
  int hh = tid & 7;
  int s  = ei[e];
  int d  = ei[NE + e];
  const float4* qp = (const float4*)(q + (size_t)d * DIM + hh * CH);
  const float4* kp = (const float4*)(k + (size_t)s * DIM + hh * CH);
  float acc = 0.0f;
#pragma unroll
  for (int i = 0; i < 4; ++i) {
    float4 a = qp[i], b = kp[i];
    acc += a.x * b.x + a.y * b.y + a.z * b.z + a.w * b.w;
  }
  acc *= 0.25f;                                          // 1/sqrt(C), C=16
  sc[tid] = acc;
  atomicMaxF(&m[d * NH + hh], acc);
}

// ---------------------------------------------------------------------------
// 5) ex = exp(score - m[dst]); denom[dst] += ex   (sc overwritten with ex)
// ---------------------------------------------------------------------------
__global__ void edge_exp_kernel(const int* __restrict__ ei,
                                float* __restrict__ sc,
                                const float* __restrict__ m,
                                float* __restrict__ den) {
  int tid = blockIdx.x * blockDim.x + threadIdx.x;      // E*H threads
  if (tid >= NE * NH) return;
  int e  = tid >> 3;
  int hh = tid & 7;
  int d  = ei[NE + e];
  float ex = __expf(sc[tid] - m[d * NH + hh]);
  sc[tid] = ex;
  atomicAdd(&den[d * NH + hh], ex);
}

// ---------------------------------------------------------------------------
// 6) agg[dst] += v[src] * (ex / (denom[dst] + 1e-16)), float4-vectorized
// ---------------------------------------------------------------------------
__global__ void edge_agg_kernel(const int* __restrict__ ei,
                                const float* __restrict__ sc,
                                const float* __restrict__ den,
                                const float* __restrict__ v,
                                float* __restrict__ agg) {
  int tid = blockIdx.x * blockDim.x + threadIdx.x;      // E*32 threads
  if (tid >= NE * 32) return;
  int e  = tid >> 5;
  int d4 = (tid & 31) << 2;     // 0,4,...,124
  int hh = d4 >> 4;
  int s  = ei[e];
  int dd = ei[NE + e];
  float alpha = sc[e * NH + hh] / (den[dd * NH + hh] + 1e-16f);
  float4 vv = *(const float4*)(v + (size_t)s * DIM + d4);
  float* ap = agg + (size_t)dd * DIM + d4;
  atomicAdd(ap + 0, vv.x * alpha);
  atomicAdd(ap + 1, vv.y * alpha);
  atomicAdd(ap + 2, vv.z * alpha);
  atomicAdd(ap + 3, vv.w * alpha);
}

// ---------------------------------------------------------------------------
// 7) gate: beta = sigmoid([agg, xr, agg-xr] @ Wb + bb)
// ---------------------------------------------------------------------------
__global__ void beta_kernel(const float* __restrict__ agg,
                            const float* __restrict__ xr,
                            const float* __restrict__ Wb,
                            const float* __restrict__ bb,
                            float* __restrict__ beta) {
  int n = blockIdx.x * blockDim.x + threadIdx.x;        // N threads
  if (n >= NTOT) return;
  float s = bb[0];
#pragma unroll 4
  for (int d = 0; d < DIM; ++d) {
    float a = agg[n * DIM + d];
    float x = xr[n * DIM + d];
    s += a * Wb[d] + x * Wb[DIM + d] + (a - x) * Wb[2 * DIM + d];
  }
  beta[n] = 1.0f / (1.0f + __expf(-s));
}

// ---------------------------------------------------------------------------
// 8) residual + LayerNorm, one node per wave32 (4 elems/lane, shfl reduce)
// ---------------------------------------------------------------------------
__global__ void resid_ln_kernel(float* __restrict__ h,
                                const float* __restrict__ xr,
                                const float* __restrict__ agg,
                                const float* __restrict__ beta,
                                const float* __restrict__ g,
                                const float* __restrict__ b) {
  const int lane = threadIdx.x & 31;
  const int wave = threadIdx.x >> 5;
  const int n = blockIdx.x * 4 + wave;
  if (n >= NTOT) return;
  const float bt = beta[n];
  float vals[4], s = 0.0f, s2 = 0.0f;
#pragma unroll
  for (int i = 0; i < 4; ++i) {
    int d = lane + 32 * i;
    float hn = bt * xr[n * DIM + d] + (1.0f - bt) * agg[n * DIM + d];
    float t = h[n * DIM + d] + hn;
    vals[i] = t; s += t; s2 += t * t;
  }
#pragma unroll
  for (int off = 16; off > 0; off >>= 1) {
    s  += __shfl_xor(s,  off, 32);
    s2 += __shfl_xor(s2, off, 32);
  }
  const float mu  = s * (1.0f / DIM);
  const float var = s2 * (1.0f / DIM) - mu * mu;
  const float inv = rsqrtf(var + 1e-5f);
#pragma unroll
  for (int i = 0; i < 4; ++i) {
    int d = lane + 32 * i;
    h[n * DIM + d] = (vals[i] - mu) * inv * g[d] + b[d];
  }
}

// ---------------------------------------------------------------------------
// 9) output head: relu(h @ Wo1 + bo1) @ Wo2 + bo2, one node per thread
// ---------------------------------------------------------------------------
__global__ void out_head_kernel(const float* __restrict__ h,
                                const float* __restrict__ Wo1,
                                const float* __restrict__ bo1,
                                const float* __restrict__ Wo2,
                                const float* __restrict__ bo2,
                                float* __restrict__ out) {
  int n = blockIdx.x * blockDim.x + threadIdx.x;
  if (n >= NTOT) return;
  float acc[64];
#pragma unroll
  for (int j = 0; j < 64; ++j) acc[j] = bo1[j];
  for (int d = 0; d < DIM; ++d) {
    float hv = h[n * DIM + d];
    const float* w = Wo1 + d * 64;
#pragma unroll
    for (int j = 0; j < 64; ++j) acc[j] += hv * w[j];
  }
  float o0 = bo2[0], o1 = bo2[1], o2 = bo2[2];
#pragma unroll
  for (int j = 0; j < 64; ++j) {
    float r = fmaxf(acc[j], 0.0f);
    o0 += r * Wo2[j * 3 + 0];
    o1 += r * Wo2[j * 3 + 1];
    o2 += r * Wo2[j * 3 + 2];
  }
  out[n * 3 + 0] = o0;
  out[n * 3 + 1] = o1;
  out[n * 3 + 2] = o2;
}

// ---------------------------------------------------------------------------
extern "C" void kernel_launch(void* const* d_in, const int* in_sizes, int n_in,
                              void* d_out, int out_size, void* d_ws, size_t ws_size,
                              hipStream_t stream) {
  const float* x    = (const float*)d_in[0];
  const int*   ei   = (const int*)  d_in[1];
  const float* pe   = (const float*)d_in[2];
  const float* Win  = (const float*)d_in[3];
  const float* bin  = (const float*)d_in[4];
  const float* Wq   = (const float*)d_in[5];
  const float* bq   = (const float*)d_in[6];
  const float* Wk   = (const float*)d_in[7];
  const float* bk   = (const float*)d_in[8];
  const float* Wv   = (const float*)d_in[9];
  const float* bv   = (const float*)d_in[10];
  const float* Ws   = (const float*)d_in[11];
  const float* bs   = (const float*)d_in[12];
  const float* Wb   = (const float*)d_in[13];
  const float* bb   = (const float*)d_in[14];
  const float* ln_g = (const float*)d_in[15];
  const float* ln_b = (const float*)d_in[16];
  const float* Wo1  = (const float*)d_in[17];
  const float* bo1  = (const float*)d_in[18];
  const float* Wo2  = (const float*)d_in[19];
  const float* bo2  = (const float*)d_in[20];

  // ---- scratch layout (~64 MB; fits inside the 192 MB L2) ----
  _Float16* h16 = (_Float16*)d_ws;                     // N*D f16 (A fragments)
  _Float16* w16 = h16 + (size_t)NTOT * DIM;            // NL*65536 f16 (B fragments)
  float* fbase  = (float*)(w16 + (size_t)NL * WFRAG_PER_LAYER);
  float* h    = fbase;                                  // N*D
  float* q    = h    + (size_t)NTOT * DIM;
  float* k    = q    + (size_t)NTOT * DIM;
  float* v    = k    + (size_t)NTOT * DIM;
  float* xr   = v    + (size_t)NTOT * DIM;
  float* agg  = xr   + (size_t)NTOT * DIM;
  float* sc   = agg  + (size_t)NTOT * DIM;              // E*H
  float* m    = sc   + (size_t)NE * NH;                 // N*H
  float* den  = m    + (size_t)NTOT * NH;               // N*H
  float* beta = den  + (size_t)NTOT * NH;               // N

  const int ND_blocks = (NTOT * DIM) / 256;  // 8192
  const int EH_blocks = (NE * NH) / 256;     // 8192
  const int E4_blocks = (NE * 32) / 256;     // 32768
  const int WP_blocks = (NL * WFRAG_PER_LAYER) / 256;   // 1536

  input_proj_kernel<<<ND_blocks, 256, 0, stream>>>(x, pe, Win, bin, h);
  pack_w16_kernel<<<WP_blocks, 256, 0, stream>>>(Wq, Wk, Wv, Ws, w16);

  for (int l = 0; l < NL; ++l) {
    pack_h16_kernel<<<ND_blocks, 256, 0, stream>>>(h, h16);
    layer_init_kernel<<<ND_blocks, 256, 0, stream>>>(agg, m, den);

    qkvs_gemm_wmma<<<1024, 256, 0, stream>>>(
        h16, w16 + (size_t)l * WFRAG_PER_LAYER,
        bq + (size_t)l * DIM, bk + (size_t)l * DIM,
        bv + (size_t)l * DIM, bs + (size_t)l * DIM,
        q, k, v, xr);

    edge_score_kernel<<<EH_blocks, 256, 0, stream>>>(ei, q, k, sc, m);
    edge_exp_kernel  <<<EH_blocks, 256, 0, stream>>>(ei, sc, m, den);
    edge_agg_kernel  <<<E4_blocks, 256, 0, stream>>>(ei, sc, den, v, agg);

    beta_kernel<<<NTOT / 256, 256, 0, stream>>>(
        agg, xr, Wb + (size_t)l * 3 * DIM, bb + l, beta);

    resid_ln_kernel<<<NTOT / 4, 128, 0, stream>>>(
        h, xr, agg, beta, ln_g + (size_t)l * DIM, ln_b + (size_t)l * DIM);
  }

  out_head_kernel<<<NTOT / 256, 256, 0, stream>>>(h, Wo1, bo1, Wo2, bo2,
                                                  (float*)d_out);
}
